// Wave2D_54331336295014
// MI455X (gfx1250) — matrix operand order repaired and checked by reading, not compiled, not run on hardware
//
#include <hip/hip_runtime.h>
#include <math.h>

// ---------------------------------------------------------------------------
// Wave2D for MI455X (gfx1250): bf16 WMMA (16x16x32, f32 accum) on all GEMM
// paths, fp32 for LayerNorm statistics and final output.  DCT GEMM stages B
// tiles into LDS via GLOBAL_LOAD_ASYNC_TO_LDS (ASYNCcnt) when the toolchain
// declares the builtin, else via vectorized register double-buffering.
// ---------------------------------------------------------------------------

typedef __bf16 bf16_t;
typedef __bf16 v16bf __attribute__((ext_vector_type(16)));
typedef float  v8f   __attribute__((ext_vector_type(8)));

// Pointer element types required by the async-to-LDS builtins (from clang's
// diagnostic: "__vector_size__(4 * sizeof(int)) int *").
typedef int wv4i __attribute__((vector_size(16)));
typedef int wv2i __attribute__((vector_size(8)));

#if defined(__has_builtin)
#if __has_builtin(__builtin_amdgcn_global_load_async_to_lds_b128) && \
    __has_builtin(__builtin_amdgcn_global_load_async_to_lds_b64)
#define WAVE2D_ASYNC_LDS 1
#endif
#endif
#ifndef WAVE2D_ASYNC_LDS
#define WAVE2D_ASYNC_LDS 0
#endif

static __device__ inline void wave2d_wait_async0() {
#if WAVE2D_ASYNC_LDS
#if __has_builtin(__builtin_amdgcn_s_wait_asynccnt)
  __builtin_amdgcn_s_wait_asynccnt(0);
#else
  asm volatile("s_wait_asynccnt 0" ::: "memory");
#endif
#endif
}

union BFrag { v16bf v; bf16_t e[16]; };
union CFrag { v8f   v; float  e[8];  };

static __device__ inline v8f wmma_bf16(v16bf a, v16bf b, v8f c) {
  // (neg_a, A, neg_b, B, c_mod, C, reuse_a, reuse_b)
  return __builtin_amdgcn_wmma_f32_16x16x32_bf16(false, a, false, b, (short)0, c,
                                                 false, false);
}

static __device__ inline v8f zero_v8f() {
  v8f z = {0.f, 0.f, 0.f, 0.f, 0.f, 0.f, 0.f, 0.f};
  return z;
}

// A-matrix fragment (16x32 bf16, MxK), source row-major [16][ld].
// Layout: lane m=lane&15; K chunk = (lane>>4)*16 + i (matches B loaders).
static __device__ inline v16bf load_a_rm(const bf16_t* A, int ld, int lane) {
  const int m  = lane & 15;
  const int kb = (lane >> 4) * 16;
  BFrag f;
#pragma unroll
  for (int i = 0; i < 16; ++i) f.e[i] = A[m * ld + kb + i];
  return f.v;
}

// B-matrix fragment (32x16 bf16, KxN) from a WEIGHT stored row-major [N][K]
// (i.e. B[k][n] = W[n][k]).  Lane col n=lane&15; K chunk = (lane>>4)*16 + i.
static __device__ inline v16bf load_b_weight(const bf16_t* W, int ld, int lane) {
  const int n  = lane & 15;
  const int kb = (lane >> 4) * 16;
  BFrag f;
#pragma unroll
  for (int i = 0; i < 16; ++i) f.e[i] = W[n * ld + kb + i];
  return f.v;
}

// B-matrix fragment from K-major storage [32][ld] (k rows, n cols).
static __device__ inline v16bf load_b_kmajor(const bf16_t* B, int ld, int lane) {
  const int n  = lane & 15;
  const int kb = (lane >> 4) * 16;
  BFrag f;
#pragma unroll
  for (int i = 0; i < 16; ++i) f.e[i] = B[(kb + i) * ld + n];
  return f.v;
}

// ---------------------------------------------------------------------------
// Kernel 1: tables — bf16 DCT basis (and transpose) + bf16 weight copies
// ---------------------------------------------------------------------------
__global__ void wave2d_build_tables(const float* __restrict__ lin_w,
                                    const float* __restrict__ tok_w,
                                    const float* __restrict__ out_w,
                                    bf16_t* __restrict__ WHb,
                                    bf16_t* __restrict__ WHTb,
                                    bf16_t* __restrict__ linb,
                                    bf16_t* __restrict__ tokb,
                                    bf16_t* __restrict__ outb) {
  const int i = blockIdx.x * 256 + threadIdx.x;
  if (i < 128 * 128) {
    const int k = i >> 7, h = i & 127;
    // cos(k*(h+0.5)*pi/128) * sqrt(2/128); row 0 scaled by 1/sqrt(2)
    float w = cosf((float)k * ((float)h + 0.5f) * 0.0245436926f) * 0.125f;
    if (k == 0) w *= 0.70710678f;
    WHb[k * 128 + h]  = (bf16_t)w;
    WHTb[h * 128 + k] = (bf16_t)w;
  }
  if (i < 192 * 96) linb[i] = (bf16_t)lin_w[i];
  if (i < 96 * 96) {
    tokb[i] = (bf16_t)tok_w[i];
    outb[i] = (bf16_t)out_w[i];
  }
}

// ---------------------------------------------------------------------------
// Kernel 2: depthwise 3x3 conv fused with C->2C linear.
// Block = 32 pixels of one row (b,h,w0..w0+31). WMMA: M=32, N=192, K=96.
// Writes x_part (bf16) and gate = silu(z) (bf16), layout [B*H*W][96].
// ---------------------------------------------------------------------------
__global__ __launch_bounds__(256) void wave2d_conv_lin(
    const float* __restrict__ x, const float* __restrict__ dw_w,
    const float* __restrict__ dw_b, const bf16_t* __restrict__ lin_wb,
    const float* __restrict__ lin_b, bf16_t* __restrict__ xpart,
    bf16_t* __restrict__ gate) {
  __shared__ __align__(16) bf16_t sA[32 * 96];  // [pixel][c_in]
  const int tid = threadIdx.x, lane = tid & 31, wv = tid >> 5;
  const int w0 = blockIdx.x * 32, h = blockIdx.y, b = blockIdx.z;

  // depthwise conv: i -> (c = i/32, wp = i%32); coalesced over wp
  for (int i = tid; i < 96 * 32; i += 256) {
    const int c = i >> 5, wp = i & 31;
    const int w = w0 + wp;
    float acc = dw_b[c];
    const float* xp = x + ((size_t)(b * 96 + c)) * 128 * 128;
#pragma unroll
    for (int dy = -1; dy <= 1; ++dy) {
      const int hy = h + dy;
      if (hy < 0 || hy >= 128) continue;
#pragma unroll
      for (int dx = -1; dx <= 1; ++dx) {
        const int wx = w + dx;
        if (wx < 0 || wx >= 128) continue;
        acc += xp[hy * 128 + wx] * dw_w[c * 9 + (dy + 1) * 3 + (dx + 1)];
      }
    }
    sA[wp * 96 + c] = (bf16_t)acc;
  }
  __syncthreads();

  const int m0 = (wv & 1) * 16;     // pixel tile
  const int jbase = wv >> 1;        // n-tiles jbase, jbase+4, jbase+8
  v8f acc[3];
#pragma unroll
  for (int t = 0; t < 3; ++t) acc[t] = zero_v8f();

#pragma unroll
  for (int kc = 0; kc < 3; ++kc) {
    const v16bf a = load_a_rm(sA + m0 * 96 + kc * 32, 96, lane);
#pragma unroll
    for (int t = 0; t < 3; ++t) {
      const int j = jbase + 4 * t;
      const v16bf bb = load_b_weight(lin_wb + j * 16 * 96 + kc * 32, 96, lane);
      acc[t] = wmma_bf16(a, bb, acc[t]);
    }
  }

  const size_t pixbase = ((size_t)(b * 128 + h)) * 128 + w0;
#pragma unroll
  for (int t = 0; t < 3; ++t) {
    const int j = jbase + 4 * t;
    const int n = j * 16 + (lane & 15);
    const float bias = lin_b[n];
    CFrag cf; cf.v = acc[t];
#pragma unroll
    for (int r = 0; r < 8; ++r) {
      const int m = m0 + (lane >> 4) * 8 + r;
      const size_t row = (pixbase + m) * 96;
      const float v = cf.e[r] + bias;
      if (n < 96) {
        xpart[row + n] = (bf16_t)v;
      } else {
        const float s = v / (1.f + expf(-v));  // silu
        gate[row + (n - 96)] = (bf16_t)s;
      }
    }
  }
}

// ---------------------------------------------------------------------------
// Kernel 3: batched DCT GEMM.  D[bz,m,n] = sum_k W[m,k] * S[bz,k,n]
// M=K=128; ld = N, batch stride = 128*N.  W basis in LDS; B K-chunks
// double-buffered in LDS (async DMA when available).
// ---------------------------------------------------------------------------
template <int NT>
__global__ __launch_bounds__(256) void wave2d_dct_gemm(
    const bf16_t* __restrict__ Wm, const bf16_t* __restrict__ S,
    bf16_t* __restrict__ Db, float* __restrict__ Df, int N, int use_f32) {
  __shared__ __align__(16) bf16_t sW[128 * 128];
  __shared__ __align__(16) bf16_t sB[2][32 * NT];
  const int tid = threadIdx.x, lane = tid & 31, wv = tid >> 5;
  const int n0 = blockIdx.x * 128;
  const size_t base = (size_t)blockIdx.y * 128 * (size_t)N;
  constexpr int NTILES = NT / 16;
  constexpr int CH  = (NT == 128) ? 8 : 4;      // bf16 per copy chunk
  constexpr int CNT = (32 * NT) / (256 * CH);   // chunks per thread
  constexpr int CW  = CH / 2;                   // dwords per chunk

  // --- stage basis (32 KB) + K-chunk 0 -------------------------------------
#if WAVE2D_ASYNC_LDS
#pragma unroll
  for (int t = 0; t < 8; ++t) {
    const int e0 = (tid + 256 * t) * 8;
    __builtin_amdgcn_global_load_async_to_lds_b128(
        (wv4i*)(Wm + e0), (wv4i*)(sW + e0), 0, 0);
  }
#pragma unroll
  for (int t = 0; t < CNT; ++t) {
    const int e0 = (tid + 256 * t) * CH;
    const int r = e0 / NT, cc = e0 - r * NT;
    const bf16_t* g = S + base + (size_t)r * N + n0 + cc;
    if (CH == 8)
      __builtin_amdgcn_global_load_async_to_lds_b128(
          (wv4i*)g, (wv4i*)(&sB[0][e0]), 0, 0);
    else
      __builtin_amdgcn_global_load_async_to_lds_b64(
          (wv2i*)g, (wv2i*)(&sB[0][e0]), 0, 0);
  }
  wave2d_wait_async0();
#else
#pragma unroll
  for (int t = 0; t < 8; ++t) {
    const int e0 = (tid + 256 * t) * 8;
    *(uint4*)(sW + e0) = *(const uint4*)(Wm + e0);
  }
#pragma unroll
  for (int t = 0; t < CNT; ++t) {
    const int e0 = (tid + 256 * t) * CH;
    const int r = e0 / NT, cc = e0 - r * NT;
    const uint32_t* g = (const uint32_t*)(S + base + (size_t)r * N + n0 + cc);
    uint32_t* l = (uint32_t*)&sB[0][e0];
#pragma unroll
    for (int w2 = 0; w2 < CW; ++w2) l[w2] = g[w2];
  }
#endif
  __syncthreads();

  v8f acc[NTILES];
#pragma unroll
  for (int j = 0; j < NTILES; ++j) acc[j] = zero_v8f();

  const int m0 = wv * 16;
#if !WAVE2D_ASYNC_LDS
  uint32_t regs[CNT][CW];
#endif

#pragma unroll
  for (int kc = 0; kc < 4; ++kc) {
    const int cur = kc & 1;
    // issue staging of next K-chunk (overlaps with WMMA below)
    if (kc < 3) {
#pragma unroll
      for (int t = 0; t < CNT; ++t) {
        const int e0 = (tid + 256 * t) * CH;
        const int r = e0 / NT, cc = e0 - r * NT;
        const bf16_t* g =
            S + base + (size_t)(32 * (kc + 1) + r) * N + n0 + cc;
#if WAVE2D_ASYNC_LDS
        if (CH == 8)
          __builtin_amdgcn_global_load_async_to_lds_b128(
              (wv4i*)g, (wv4i*)(&sB[cur ^ 1][e0]), 0, 0);
        else
          __builtin_amdgcn_global_load_async_to_lds_b64(
              (wv2i*)g, (wv2i*)(&sB[cur ^ 1][e0]), 0, 0);
#else
        const uint32_t* gw = (const uint32_t*)g;
#pragma unroll
        for (int w2 = 0; w2 < CW; ++w2) regs[t][w2] = gw[w2];
#endif
      }
    }
    if (kc < 2) {  // speculative prefetch of the chunk after next into L2
      const int e0 = tid * CH;
      const int r = e0 / NT, cc = e0 - r * NT;
      __builtin_prefetch(S + base + (size_t)(32 * (kc + 2) + r) * N + n0 + cc,
                         0, 0);
    }

    const v16bf a = load_a_rm(sW + m0 * 128 + kc * 32, 128, lane);
#pragma unroll
    for (int j = 0; j < NTILES; ++j) {
      const v16bf bb = load_b_kmajor(sB[cur] + j * 16, NT, lane);
      acc[j] = wmma_bf16(a, bb, acc[j]);
    }

    if (kc < 3) {
#if WAVE2D_ASYNC_LDS
      wave2d_wait_async0();
#else
#pragma unroll
      for (int t = 0; t < CNT; ++t) {
        uint32_t* l = (uint32_t*)&sB[cur ^ 1][(tid + 256 * t) * CH];
#pragma unroll
        for (int w2 = 0; w2 < CW; ++w2) l[w2] = regs[t][w2];
      }
#endif
    }
    __syncthreads();
  }

#pragma unroll
  for (int j = 0; j < NTILES; ++j) {
    CFrag cf; cf.v = acc[j];
    const int n = n0 + j * 16 + (lane & 15);
#pragma unroll
    for (int r = 0; r < 8; ++r) {
      const int m = m0 + (lane >> 4) * 8 + r;
      const size_t idx = base + (size_t)m * N + n;
      if (use_f32) Df[idx] = cf.e[r];
      else         Db[idx] = (bf16_t)cf.e[r];
    }
  }
}

// ---------------------------------------------------------------------------
// Kernel 4: t = gelu(freq_embed @ tok_w^T + tok_b);
// final = u0 * (cos(ct) + sin(ct)*(1+alpha/2)/c_safe)   [since v0 == u0]
// Block = 128 pixels. WMMA: M=128, N=96, K=96.
// ---------------------------------------------------------------------------
__global__ __launch_bounds__(256) void wave2d_freq_mod(
    const float* __restrict__ fe, const bf16_t* __restrict__ tok_wb,
    const float* __restrict__ tok_b, const bf16_t* __restrict__ u0,
    bf16_t* __restrict__ finalb, const float* __restrict__ c_ptr,
    const float* __restrict__ a_ptr) {
  __shared__ __align__(16) bf16_t sA[128 * 96];
  const int tid = threadIdx.x, lane = tid & 31, wv = tid >> 5;
  const size_t pix0 = (size_t)blockIdx.x * 128;

  for (int i = tid * 4; i < 128 * 96; i += 256 * 4) {
    const float4 f = *(const float4*)(fe + pix0 * 96 + i);
    union { bf16_t b[4]; uint2 u; } o;
    o.b[0] = (bf16_t)f.x; o.b[1] = (bf16_t)f.y;
    o.b[2] = (bf16_t)f.z; o.b[3] = (bf16_t)f.w;
    *(uint2*)(sA + i) = o.u;
  }
  __syncthreads();

  const float c_safe = fabsf(c_ptr[0]) + 1e-4f;
  const float alpha_safe = fmaxf(a_ptr[0], 0.f);
  const float sfac = (1.f + 0.5f * alpha_safe) / c_safe;

  const int m0 = wv * 16;
  v8f acc[6];
#pragma unroll
  for (int j = 0; j < 6; ++j) acc[j] = zero_v8f();

#pragma unroll
  for (int kc = 0; kc < 3; ++kc) {
    const v16bf a = load_a_rm(sA + m0 * 96 + kc * 32, 96, lane);
#pragma unroll
    for (int j = 0; j < 6; ++j) {
      const v16bf bb = load_b_weight(tok_wb + j * 16 * 96 + kc * 32, 96, lane);
      acc[j] = wmma_bf16(a, bb, acc[j]);
    }
  }

#pragma unroll
  for (int j = 0; j < 6; ++j) {
    const int n = j * 16 + (lane & 15);
    const float bias = tok_b[n];
    CFrag cf; cf.v = acc[j];
#pragma unroll
    for (int r = 0; r < 8; ++r) {
      const int m = m0 + (lane >> 4) * 8 + r;
      float t = cf.e[r] + bias;
      t = 0.5f * t * (1.f + erff(t * 0.70710678f));  // exact gelu
      const float ct = fminf(fmaxf(c_safe * t, -20.f), 20.f);
      const float fac = cosf(ct) + sinf(ct) * sfac;
      const size_t idx = (pix0 + m) * 96 + n;
      finalb[idx] = (bf16_t)((float)u0[idx] * fac);
    }
  }
}

// ---------------------------------------------------------------------------
// Kernel 5: LayerNorm(C) -> *gate -> @ out_w^T + out_b -> NCHW store.
// Block = 32 pixels; 8 lanes per pixel for LN stats (wave32 shfl_xor).
// ---------------------------------------------------------------------------
__global__ __launch_bounds__(256) void wave2d_ln_gate_out(
    const float* __restrict__ xf, const bf16_t* __restrict__ gate,
    const float* __restrict__ ln_g, const float* __restrict__ ln_b,
    const bf16_t* __restrict__ out_wb, const float* __restrict__ out_b,
    float* __restrict__ out) {
  __shared__ __align__(16) bf16_t sA[32 * 96];
  const int tid = threadIdx.x, lane = tid & 31, wv = tid >> 5;
  const size_t pix0 = (size_t)blockIdx.x * 32;
  const int p = tid >> 3;   // pixel within tile
  const int cg = tid & 7;   // column group (8 lanes/pixel)

  float v[12], s = 0.f, s2 = 0.f;
  const size_t rowbase = (pix0 + p) * 96;
#pragma unroll
  for (int i = 0; i < 12; ++i) {
    const float t = xf[rowbase + cg + 8 * i];
    v[i] = t; s += t; s2 += t * t;
  }
#pragma unroll
  for (int m = 1; m < 8; m <<= 1) {
    s  += __shfl_xor(s,  m, 32);
    s2 += __shfl_xor(s2, m, 32);
  }
  const float mu = s * (1.f / 96.f);
  const float var = s2 * (1.f / 96.f) - mu * mu;
  const float rs = rsqrtf(var + 1e-5f);
#pragma unroll
  for (int i = 0; i < 12; ++i) {
    const int c = cg + 8 * i;
    const float xn = (v[i] - mu) * rs * ln_g[c] + ln_b[c];
    sA[p * 96 + c] = (bf16_t)(xn * (float)gate[rowbase + c]);
  }
  __syncthreads();

  const int m0 = (wv & 1) * 16;
  for (int j = wv >> 1; j < 6; j += 4) {
    v8f acc = zero_v8f();
#pragma unroll
    for (int kc = 0; kc < 3; ++kc) {
      const v16bf a = load_a_rm(sA + m0 * 96 + kc * 32, 96, lane);
      const v16bf bb = load_b_weight(out_wb + j * 16 * 96 + kc * 32, 96, lane);
      acc = wmma_bf16(a, bb, acc);
    }
    const int n = j * 16 + (lane & 15);
    const float bias = out_b[n];
    CFrag cf; cf.v = acc;
#pragma unroll
    for (int r = 0; r < 8; ++r) {
      const int m = m0 + (lane >> 4) * 8 + r;
      const size_t pix = pix0 + m;
      const size_t b = pix >> 14;        // / 16384
      const size_t hw = pix & 16383;
      out[(b * 96 + n) * 16384 + hw] = cf.e[r] + bias;
    }
  }
}

// ---------------------------------------------------------------------------
// Host launcher
// ---------------------------------------------------------------------------
extern "C" void kernel_launch(void* const* d_in, const int* in_sizes, int n_in,
                              void* d_out, int out_size, void* d_ws,
                              size_t ws_size, hipStream_t stream) {
  (void)in_sizes; (void)n_in; (void)out_size; (void)ws_size;

  const float* x      = (const float*)d_in[0];
  const float* fe     = (const float*)d_in[1];
  const float* dw_w   = (const float*)d_in[2];
  const float* dw_b   = (const float*)d_in[3];
  const float* lin_w  = (const float*)d_in[4];
  const float* lin_b  = (const float*)d_in[5];
  const float* tok_w  = (const float*)d_in[6];
  const float* tok_b  = (const float*)d_in[7];
  const float* ln_g   = (const float*)d_in[8];
  const float* ln_b   = (const float*)d_in[9];
  const float* out_w  = (const float*)d_in[10];
  const float* out_b  = (const float*)d_in[11];
  const float* c_s    = (const float*)d_in[12];
  const float* a_s    = (const float*)d_in[13];

  const size_t NEL = (size_t)8 * 128 * 128 * 96;  // 12,582,912
  char* ws = (char*)d_ws;
  bf16_t* xpart = (bf16_t*)ws;  ws += NEL * 2;
  bf16_t* gate  = (bf16_t*)ws;  ws += NEL * 2;
  bf16_t* tmp1  = (bf16_t*)ws;  ws += NEL * 2;
  bf16_t* tmp2  = (bf16_t*)ws;  ws += NEL * 2;
  float*  xff   = (float*)ws;   ws += NEL * 4;
  bf16_t* WHb   = (bf16_t*)ws;  ws += 128 * 128 * 2;
  bf16_t* WHTb  = (bf16_t*)ws;  ws += 128 * 128 * 2;
  bf16_t* linb  = (bf16_t*)ws;  ws += 192 * 96 * 2;
  bf16_t* tokb  = (bf16_t*)ws;  ws += 96 * 96 * 2;
  bf16_t* outb  = (bf16_t*)ws;  ws += 96 * 96 * 2;

  wave2d_build_tables<<<(192 * 96 + 255) / 256, 256, 0, stream>>>(
      lin_w, tok_w, out_w, WHb, WHTb, linb, tokb, outb);

  wave2d_conv_lin<<<dim3(4, 128, 8), 256, 0, stream>>>(
      x, dw_w, dw_b, linb, lin_b, xpart, gate);

  // DCT along H (per b: 128x128 @ 128x12288), then along W (per (b,k))
  wave2d_dct_gemm<128><<<dim3(96, 8), 256, 0, stream>>>(
      WHb, xpart, tmp1, nullptr, 12288, 0);
  wave2d_dct_gemm<96><<<dim3(1, 1024), 256, 0, stream>>>(
      WHb, tmp1, tmp2, nullptr, 96, 0);

  wave2d_freq_mod<<<1024, 256, 0, stream>>>(
      fe, tokb, tok_b, tmp2, tmp1, c_s, a_s);

  // IDCT along W (transposed basis), then along H -> f32
  wave2d_dct_gemm<96><<<dim3(1, 1024), 256, 0, stream>>>(
      WHTb, tmp1, tmp2, nullptr, 96, 0);
  wave2d_dct_gemm<128><<<dim3(96, 8), 256, 0, stream>>>(
      WHTb, tmp2, nullptr, xff, 12288, 1);

  wave2d_ln_gate_out<<<4096, 256, 0, stream>>>(
      xff, gate, ln_g, ln_b, outb, out_b, (float*)d_out);
}
